// LightweightMultiDirectionalMAMBADiffusion_18433999634576
// MI455X (gfx1250) — matrix operand order repaired and checked by reading, not compiled, not run on hardware
//
#include <hip/hip_runtime.h>
#include <math.h>

// ---------------------------------------------------------------------------
// Model dims (match reference)
// ---------------------------------------------------------------------------
#define L_SEQ   1024
#define N_IN    512
#define N_OUT   512
#define DMODEL  256
#define DINNER  512
#define DSTATE  16
#define NLAYERS 6
#define NHEADS  8
#define HDIM    32
#define NFREQ   256
#define FEAT    515      // 2*NFREQ + 3 (concat of fourier feats + values)
#define LNEPS   1e-5f

typedef _Float16 v16h __attribute__((ext_vector_type(16)));
typedef float    v8f  __attribute__((ext_vector_type(8)));

union F16Frag { v16h v; unsigned u[8]; };

enum { ACT_NONE = 0, ACT_GELU = 1, ACT_SILU = 2, ACT_SOFTPLUS = 3 };

__device__ __forceinline__ float act_apply(float x, int act) {
  switch (act) {
    case ACT_GELU:     return 0.5f * x * (1.0f + erff(x * 0.70710678118654752f));
    case ACT_SILU:     return x / (1.0f + __expf(-x));
    case ACT_SOFTPLUS: return (x > 20.0f) ? x : log1pf(__expf(x));
  }
  return x;
}

// ---------------------------------------------------------------------------
// WMMA GEMM:  C[M,N] = act( A[M,K] @ W[K,N] + bias )
// 128x64 block tile, 256 threads = 8 wave32 waves (4 Mtiles x 2 Nstrips);
// each wave owns a 32x32 output tile -> 2 A-frags x 2 B-frags reused across
// four v_wmma_f32_16x16x32_f16 per K-step (8 ds loads per WMMA).
// Double-buffered LDS with register prefetch of the next K-tile; interior
// tiles staged with float4 (global_load_b128), edges with clamped
// mask-multiply scalar loads (branch-free: no exec-mask divergence).
// ---------------------------------------------------------------------------
#define LDK 36   // 32 k-halves + 4 pad; 72B rows -> quad (8B) aligned ds stores

__global__ __launch_bounds__(256)
void k_gemm(const float* __restrict__ A, const float* __restrict__ W,
            const float* __restrict__ bias, float* __restrict__ C,
            int M, int N, int K, int act) {
  __shared__ _Float16 lA[2][128 * LDK];   // [m][k]
  __shared__ _Float16 lB[2][64 * LDK];    // [n][k]  (transposed stage)

  const int tid  = threadIdx.x;
  const int lane = tid & 31;
  const int wid  = tid >> 5;
  const int wm   = wid >> 1;        // 0..3 (32-row M tile)
  const int wn   = wid & 1;         // 0..1 (32-col N tile)
  const int hs   = (lane >> 4) & 1; // half-wave select
  const int lr   = lane & 15;

  const int bm = blockIdx.y * 128;
  const int bn = blockIdx.x * 64;

  const bool fastA_dim = ((K & 3) == 0) && (bm + 128 <= M);
  const bool fastB_dim = ((N & 3) == 0) && (bn + 64 <= N);

  float a_reg[16], b_reg[8];

  // ---- branch-light tile loader into registers -------------------------
  auto load_tiles = [&](int k0) {
    const bool kin = (k0 + 32 <= K);
#pragma unroll
    for (int i = 0; i < 4; ++i) {   // A: 128 rows x 32 k ; quad -> (r, c4)
      int q = tid + i * 256;
      int r = q >> 3, c4 = (q & 7) << 2;
      int gm = bm + r;
      if (fastA_dim && kin) {
        const float4 v = *reinterpret_cast<const float4*>(A + (size_t)gm * K + k0 + c4);
        a_reg[i * 4 + 0] = v.x; a_reg[i * 4 + 1] = v.y;
        a_reg[i * 4 + 2] = v.z; a_reg[i * 4 + 3] = v.w;
      } else {
#pragma unroll
        for (int j = 0; j < 4; ++j) {
          int gk = k0 + c4 + j;
          bool ok = (gm < M) && (gk < K);
          size_t idx = ok ? ((size_t)gm * K + gk) : 0;
          float v = A[idx];
          a_reg[i * 4 + j] = ok ? v : 0.0f;
        }
      }
    }
#pragma unroll
    for (int i = 0; i < 2; ++i) {   // B: 32 k x 64 n ; quad -> (kk, n4)
      int q = tid + i * 256;
      int kk = q >> 4, n4 = (q & 15) << 2;
      int gk = k0 + kk;
      if (fastB_dim && kin) {
        const float4 v = *reinterpret_cast<const float4*>(W + (size_t)gk * N + bn + n4);
        b_reg[i * 4 + 0] = v.x; b_reg[i * 4 + 1] = v.y;
        b_reg[i * 4 + 2] = v.z; b_reg[i * 4 + 3] = v.w;
      } else {
#pragma unroll
        for (int j = 0; j < 4; ++j) {
          int gn = bn + n4 + j;
          bool ok = (gk < K) && (gn < N);
          size_t idx = ok ? ((size_t)gk * N + gn) : 0;
          float v = W[idx];
          b_reg[i * 4 + j] = ok ? v : 0.0f;
        }
      }
    }
  };

  auto store_stage = [&](int p) {
#pragma unroll
    for (int i = 0; i < 4; ++i) {   // A: one aligned 8B ds store per quad
      int q = tid + i * 256;
      int r = q >> 3, c4 = (q & 7) << 2;
      union { _Float16 h[4]; uint2 u2; } pk;
#pragma unroll
      for (int j = 0; j < 4; ++j) pk.h[j] = (_Float16)a_reg[i * 4 + j];
      *reinterpret_cast<uint2*>(&lA[p][r * LDK + c4]) = pk.u2;
    }
#pragma unroll
    for (int i = 0; i < 2; ++i) {   // B: transposed scatter (4 x b16)
      int q = tid + i * 256;
      int kk = q >> 4, n4 = (q & 15) << 2;
#pragma unroll
      for (int j = 0; j < 4; ++j)
        lB[p][(n4 + j) * LDK + kk] = (_Float16)b_reg[i * 4 + j];
    }
  };

  v8f acc00 = {0.f,0.f,0.f,0.f,0.f,0.f,0.f,0.f};
  v8f acc01 = acc00, acc10 = acc00, acc11 = acc00;

  const int nk = (K + 31) >> 5;
  load_tiles(0);
  int p = 0;
  for (int t = 0; t < nk; ++t) {
    store_stage(p);
    __syncthreads();
    if (t + 1 < nk) load_tiles((t + 1) << 5);   // prefetch next tile into regs

    // ---- fragments per ISA 16-bit layouts, from buffer p ----
    F16Frag A0, A1, B0, B1;
    const _Float16* ar0 = &lA[p][(wm * 32 + lr) * LDK];
    const _Float16* ar1 = &lA[p][(wm * 32 + 16 + lr) * LDK];
    const _Float16* br0 = &lB[p][(wn * 32 + lr) * LDK];
    const _Float16* br1 = &lB[p][(wn * 32 + 16 + lr) * LDK];
#pragma unroll
    for (int q = 0; q < 8; ++q) {
      int ka = ((q < 4) ? (2 * q) : (16 + 2 * (q - 4))) + (hs ? 8 : 0);
      A0.u[q] = *reinterpret_cast<const unsigned*>(ar0 + ka);
      A1.u[q] = *reinterpret_cast<const unsigned*>(ar1 + ka);
      int kb = ((q < 4) ? (2 * q) : (16 + 2 * (q - 4))) + (hs ? 8 : 0);
      B0.u[q] = *reinterpret_cast<const unsigned*>(br0 + kb);
      B1.u[q] = *reinterpret_cast<const unsigned*>(br1 + kb);
    }
    acc00 = __builtin_amdgcn_wmma_f32_16x16x32_f16(false, A0.v, false, B0.v, (short)0, acc00, false, false);
    acc01 = __builtin_amdgcn_wmma_f32_16x16x32_f16(false, A0.v, false, B1.v, (short)0, acc01, false, false);
    acc10 = __builtin_amdgcn_wmma_f32_16x16x32_f16(false, A1.v, false, B0.v, (short)0, acc10, false, false);
    acc11 = __builtin_amdgcn_wmma_f32_16x16x32_f16(false, A1.v, false, B1.v, (short)0, acc11, false, false);
    p ^= 1;
  }

  // ---- epilogue: bias + activation, f32 store ----
  int n0 = bn + wn * 32 + lr;
  int n1 = n0 + 16;
  float bz0 = (bias && n0 < N) ? bias[n0] : 0.0f;
  float bz1 = (bias && n1 < N) ? bias[n1] : 0.0f;
#pragma unroll
  for (int r = 0; r < 8; ++r) {
    int m0 = bm + wm * 32 + r + hs * 8;
    int m1 = m0 + 16;
    if (m0 < M) {
      if (n0 < N) C[(size_t)m0 * N + n0] = act_apply(acc00[r] + bz0, act);
      if (n1 < N) C[(size_t)m0 * N + n1] = act_apply(acc01[r] + bz1, act);
    }
    if (m1 < M) {
      if (n0 < N) C[(size_t)m1 * N + n0] = act_apply(acc10[r] + bz0, act);
      if (n1 < N) C[(size_t)m1 * N + n1] = act_apply(acc11[r] + bz1, act);
    }
  }
}

// ---------------------------------------------------------------------------
// Sinusoidal time embedding (t_emb0[256])
// ---------------------------------------------------------------------------
__global__ void k_temb(const float* __restrict__ t, float* __restrict__ e) {
  int i = threadIdx.x;
  const float c = -logf(10000.0f) / 127.0f;
  float tv = t[0];
  if (i < 128) e[i] = __sinf(tv * __expf((float)i * c));
  else         e[i] = __cosf(tv * __expf((float)(i - 128) * c));
}

// ---------------------------------------------------------------------------
// Fourier features + value concat: out[row, 0:256]=sin, [256:512]=cos, [512:515]=vals
// ---------------------------------------------------------------------------
__global__ void k_tokcat(const float* __restrict__ coords, const float* __restrict__ vals,
                         const float* __restrict__ Bf, float* __restrict__ out) {
  int row = blockIdx.x;
  int j = threadIdx.x;  // 0..255
  float cx = coords[row * 2 + 0], cy = coords[row * 2 + 1];
  float proj = 6.283185307179586f * (cx * Bf[j * 2 + 0] + cy * Bf[j * 2 + 1]);
  out[(size_t)row * FEAT + j] = __sinf(proj);
  out[(size_t)row * FEAT + NFREQ + j] = __cosf(proj);
  if (j < 3) out[(size_t)row * FEAT + 2 * NFREQ + j] = vals[row * 3 + j];
}

// ---------------------------------------------------------------------------
// Broadcast add of t_emb to every row, and elementwise residual add
// ---------------------------------------------------------------------------
__global__ void k_addb(float* __restrict__ x, const float* __restrict__ e, int rows) {
  int i = blockIdx.x * blockDim.x + threadIdx.x;
  int total = rows * DMODEL;
  if (i < total) x[i] += e[i % DMODEL];
}

__global__ void k_add(float* __restrict__ a, const float* __restrict__ b, int n) {
  int i = blockIdx.x * blockDim.x + threadIdx.x;
  if (i < n) a[i] += b[i];
}

// ---------------------------------------------------------------------------
// LayerNorm over last dim D
// ---------------------------------------------------------------------------
__global__ __launch_bounds__(256)
void k_ln(const float* __restrict__ x, const float* __restrict__ g,
          const float* __restrict__ b, float* __restrict__ o, int D) {
  __shared__ float red[256];
  int row = blockIdx.x, tid = threadIdx.x;
  const float* xr = x + (size_t)row * D;

  float s = 0.f;
  for (int i = tid; i < D; i += 256) s += xr[i];
  red[tid] = s; __syncthreads();
  for (int st = 128; st > 0; st >>= 1) { if (tid < st) red[tid] += red[tid + st]; __syncthreads(); }
  float mean = red[0] / (float)D;
  __syncthreads();

  float v = 0.f;
  for (int i = tid; i < D; i += 256) { float d = xr[i] - mean; v += d * d; }
  red[tid] = v; __syncthreads();
  for (int st = 128; st > 0; st >>= 1) { if (tid < st) red[tid] += red[tid + st]; __syncthreads(); }
  float rstd = rsqrtf(red[0] / (float)D + LNEPS);

  for (int i = tid; i < D; i += 256)
    o[(size_t)row * D + i] = (xr[i] - mean) * rstd * g[i] + b[i];
}

// ---------------------------------------------------------------------------
// Bitonic lexsort producing the 4 scan orderings (blockIdx.x = direction)
//   0 row : (cy, cx)   1 col : (cx, cy)   2 diag : (cx+cy, cx)   3 anti : (cy-cx, cx)
// ---------------------------------------------------------------------------
__device__ __forceinline__ unsigned sortable_f32(float f) {
  unsigned u = __float_as_uint(f);
  return (u & 0x80000000u) ? ~u : (u | 0x80000000u);
}

__global__ __launch_bounds__(512)
void k_sort(const float* __restrict__ icoords, const float* __restrict__ qcoords,
            int* __restrict__ idxbuf) {
  __shared__ unsigned long long key[L_SEQ];
  __shared__ int sidx[L_SEQ];
  int dir = blockIdx.x;

  for (int i = threadIdx.x; i < L_SEQ; i += blockDim.x) {
    float cx = (i < N_IN) ? icoords[i * 2 + 0] : qcoords[(i - N_IN) * 2 + 0];
    float cy = (i < N_IN) ? icoords[i * 2 + 1] : qcoords[(i - N_IN) * 2 + 1];
    float prim, sec;
    if      (dir == 0) { prim = cy;      sec = cx; }
    else if (dir == 1) { prim = cx;      sec = cy; }
    else if (dir == 2) { prim = cx + cy; sec = cx; }
    else               { prim = cy - cx; sec = cx; }
    key[i] = ((unsigned long long)sortable_f32(prim) << 32) | sortable_f32(sec);
    sidx[i] = i;
  }
  __syncthreads();

  for (int ksz = 2; ksz <= L_SEQ; ksz <<= 1) {
    for (int j = ksz >> 1; j > 0; j >>= 1) {
      for (int i = threadIdx.x; i < L_SEQ; i += blockDim.x) {
        int ixj = i ^ j;
        if (ixj > i) {
          bool asc = ((i & ksz) == 0);
          bool sw = asc ? (key[i] > key[ixj]) : (key[i] < key[ixj]);
          if (sw) {
            unsigned long long tk = key[i]; key[i] = key[ixj]; key[ixj] = tk;
            int ti = sidx[i]; sidx[i] = sidx[ixj]; sidx[ixj] = ti;
          }
        }
      }
      __syncthreads();
    }
  }
  for (int i = threadIdx.x; i < L_SEQ; i += blockDim.x)
    idxbuf[dir * L_SEQ + i] = sidx[i];
}

// ---------------------------------------------------------------------------
// Selective SSM scan, one direction per blockIdx.y.
// dt/B/C computed once on the unpermuted sequence (they commute with the row
// permutation). The gathered operand rows (Bm/Cm rows + this block's dt/x
// slices at permuted row sord[t]) are prefetched chunk-by-chunk into a
// double-buffered LDS ring with CDNA5 async loads (global_load_async_to_lds),
// synchronized with s_wait_asynccnt so the next chunk stays in flight while
// the current chunk's recurrence is evaluated.
// 16 lanes (one per state n) per channel d; shuffle-reduce the C.h contraction.
// ---------------------------------------------------------------------------
#define SSM_CHUNK 64

__global__ __launch_bounds__(256)
void k_ssm(const float* __restrict__ x, const float* __restrict__ dt,
           const float* __restrict__ Bm, const float* __restrict__ Cm,
           const float* __restrict__ A_log, const float* __restrict__ Dp,
           const int* __restrict__ idxbuf, float* __restrict__ yf) {
  __shared__ int sord[L_SEQ];
  __shared__ float sb[2][SSM_CHUNK][4][16];   // [buf][t][Bm,Cm,dt,x][16]

  int dir = blockIdx.y;
  int d0  = blockIdx.x * 16;
  int ch  = threadIdx.x >> 4;   // 0..15 channel within block
  int n   = threadIdx.x & 15;   // state index
  int d   = d0 + ch;

  for (int i = threadIdx.x; i < L_SEQ; i += 256) sord[i] = idxbuf[dir * L_SEQ + i];
  __syncthreads();

  // async-stage one 64-step chunk: 4 b128 copies per thread
  auto stage = [&](int c, int buf) {
#pragma unroll
    for (int i = 0; i < 4; ++i) {
      int cq = threadIdx.x + i * 256;
      int t  = cq >> 4;
      int wh = (cq >> 2) & 3;
      int e4 = (cq & 3) << 2;
      int l  = sord[c * SSM_CHUNK + t];
      const float* g =
          (wh == 0) ? (Bm + (size_t)l * DSTATE + e4)
        : (wh == 1) ? (Cm + (size_t)l * DSTATE + e4)
        : (wh == 2) ? (dt + (size_t)l * DINNER + d0 + e4)
                    : (x  + (size_t)l * DINNER + d0 + e4);
      unsigned lds = (unsigned)(size_t)(&sb[buf][t][wh][e4]);  // LDS offset = low 32 bits
      unsigned long long ga = (unsigned long long)(size_t)g;
      asm volatile("global_load_async_to_lds_b128 %0, %1, off"
                   :: "v"(lds), "v"(ga) : "memory");
    }
  };

  float a  = -__expf(A_log[d * DSTATE + n]);
  float dp = Dp[d];
  float h  = 0.0f;

  const int nchunk = L_SEQ / SSM_CHUNK;
  stage(0, 0);
  for (int c = 0; c < nchunk; ++c) {
    int buf = c & 1;
    if (c + 1 < nchunk) {
      stage(c + 1, buf ^ 1);                              // keep next chunk in flight
      asm volatile("s_wait_asynccnt 0x4" ::: "memory");   // wait current chunk only
    } else {
      asm volatile("s_wait_asynccnt 0x0" ::: "memory");
    }
    __syncthreads();

    for (int tt = 0; tt < SSM_CHUNK; ++tt) {
      int l = sord[c * SSM_CHUNK + tt];
      float bv  = sb[buf][tt][0][n];
      float cv  = sb[buf][tt][1][n];
      float dtv = sb[buf][tt][2][ch];
      float xv  = sb[buf][tt][3][ch];
      h = __expf(dtv * a) * h + dtv * xv * bv;
      float cc = cv * h;
#pragma unroll
      for (int m = 1; m < 16; m <<= 1) cc += __shfl_xor(cc, m, 32);
      if (n == 0)
        yf[(size_t)l * (4 * DINNER) + dir * DINNER + d] = cc + dp * xv;
    }
    __syncthreads();   // protect buf before it is overwritten next iteration
  }
}

// ---------------------------------------------------------------------------
// Attention: one block per (query row, head). Two-pass softmax in LDS.
// ---------------------------------------------------------------------------
__global__ __launch_bounds__(128)
void k_attn(const float* __restrict__ q, const float* __restrict__ k,
            const float* __restrict__ v, float* __restrict__ o) {
  __shared__ float qs[HDIM];
  __shared__ float sc[N_IN];
  __shared__ float red[128];
  int h = blockIdx.y, qi = blockIdx.x, tid = threadIdx.x;

  if (tid < HDIM) qs[tid] = q[(size_t)qi * DMODEL + h * HDIM + tid];
  __syncthreads();

  float lmax = -1e30f;
  for (int kk = tid; kk < N_IN; kk += 128) {
    float dp = 0.f;
#pragma unroll
    for (int j = 0; j < HDIM; ++j) dp += qs[j] * k[(size_t)kk * DMODEL + h * HDIM + j];
    dp *= 0.17677669529663687f;  // 1/sqrt(32)
    sc[kk] = dp;
    lmax = fmaxf(lmax, dp);
  }
  red[tid] = lmax; __syncthreads();
  for (int st = 64; st > 0; st >>= 1) { if (tid < st) red[tid] = fmaxf(red[tid], red[tid + st]); __syncthreads(); }
  float mx = red[0];
  __syncthreads();

  float lsum = 0.f;
  for (int kk = tid; kk < N_IN; kk += 128) {
    float e = __expf(sc[kk] - mx);
    sc[kk] = e;
    lsum += e;
  }
  red[tid] = lsum; __syncthreads();
  for (int st = 64; st > 0; st >>= 1) { if (tid < st) red[tid] += red[tid + st]; __syncthreads(); }
  float inv = 1.0f / red[0];
  __syncthreads();

  if (tid < HDIM) {
    float acc = 0.f;
    for (int kk = 0; kk < N_IN; ++kk) acc += sc[kk] * v[(size_t)kk * DMODEL + h * HDIM + tid];
    o[(size_t)qi * DMODEL + h * HDIM + tid] = acc * inv;
  }
}

// ---------------------------------------------------------------------------
// Host side
// ---------------------------------------------------------------------------
static inline void gemm(hipStream_t s, const float* A, const float* W,
                        const float* bias, float* C, int M, int N, int K, int act) {
  dim3 g((N + 63) / 64, (M + 127) / 128), b(256);
  k_gemm<<<g, b, 0, s>>>(A, W, bias, C, M, N, K, act);
}

// d_in layout follows JAX pytree flattening (dict keys sorted; ASCII order).
// 0..4: noisy_values, query_coords, t, input_coords, input_values
// 5: attn_k.b 6: attn_k.w 7: attn_o.b 8: attn_o.w 9: attn_q.b 10: attn_q.w
// 11: attn_v.b 12: attn_v.w 13: dec1.b 14: dec1.w 15: dec2.b 16: dec2.w
// 17: dec3.b 18: dec3.w 19: dec_ln_b 20: dec_ln_g 21: fourier_B
// 22: input_proj.b 23: input_proj.w
// 24 + 20*layer: [A_log, Dp, W_B, W_C, W_dt.b, W_dt.w, fusion1.b, fusion1.w,
//                 fusion2.b, fusion2.w, ln_b, ln_g, mlp1.b, mlp1.w,
//                 mlp2.b, mlp2.w, proj_in.b, proj_in.w, proj_out.b, proj_out.w]
// 144: query_proj.b 145: query_proj.w 146..149: time_mlp1.b/w time_mlp2.b/w
extern "C" void kernel_launch(void* const* d_in, const int* in_sizes, int n_in,
                              void* d_out, int out_size, void* d_ws, size_t ws_size,
                              hipStream_t stream) {
  (void)in_sizes; (void)n_in; (void)out_size; (void)ws_size;
  const float* noisy   = (const float*)d_in[0];
  const float* qcoords = (const float*)d_in[1];
  const float* tval    = (const float*)d_in[2];
  const float* icoords = (const float*)d_in[3];
  const float* ivals   = (const float*)d_in[4];
  auto P = [&](int i) { return (const float*)d_in[i]; };

  // workspace carve-out
  float* w = (float*)d_ws;
  size_t off = 0;
  auto alloc = [&](size_t nf) { float* r = w + off; off += nf; return r; };
  float* temb0 = alloc(256);
  float* th    = alloc(256);
  float* temb  = alloc(256);
  float* incat = alloc((size_t)N_IN * FEAT);
  float* qcat  = alloc((size_t)N_OUT * FEAT);
  float* seq   = alloc((size_t)L_SEQ * DMODEL);
  float* xin   = alloc((size_t)L_SEQ * DINNER);
  float* dtb   = alloc((size_t)L_SEQ * DINNER);
  float* Bm    = alloc((size_t)L_SEQ * DSTATE);
  float* Cm    = alloc((size_t)L_SEQ * DSTATE);
  float* yf    = alloc((size_t)L_SEQ * 4 * DINNER);
  float* f1    = alloc((size_t)L_SEQ * 2 * DINNER);
  float* f2    = alloc((size_t)L_SEQ * DINNER);
  float* t256  = alloc((size_t)L_SEQ * DMODEL);
  float* hln   = alloc((size_t)L_SEQ * DMODEL);
  float* mh    = alloc((size_t)L_SEQ * 4 * DMODEL);
  float* qb    = alloc((size_t)N_OUT * DMODEL);
  float* kb    = alloc((size_t)N_IN * DMODEL);
  float* vb    = alloc((size_t)N_IN * DMODEL);
  float* ao    = alloc((size_t)N_OUT * DMODEL);
  float* attn  = alloc((size_t)N_OUT * DMODEL);
  float* dh1   = alloc((size_t)N_OUT * 2 * DMODEL);
  int*   idxb  = (int*)alloc(4 * L_SEQ);

  // --- time embedding -> time MLP ---
  k_temb<<<1, 256, 0, stream>>>(tval, temb0);
  gemm(stream, temb0, P(147), P(146), th, 1, DMODEL, DMODEL, ACT_SILU);   // time_mlp1 + silu
  gemm(stream, th, P(149), P(148), temb, 1, DMODEL, DMODEL, ACT_NONE);    // time_mlp2

  // --- tokens ---
  k_tokcat<<<N_IN, 256, 0, stream>>>(icoords, ivals, P(21), incat);
  k_tokcat<<<N_OUT, 256, 0, stream>>>(qcoords, noisy, P(21), qcat);
  gemm(stream, incat, P(23), P(22), seq, N_IN, DMODEL, FEAT, ACT_NONE);                     // input_proj
  gemm(stream, qcat, P(145), P(144), seq + (size_t)N_IN * DMODEL, N_OUT, DMODEL, FEAT, ACT_NONE); // query_proj
  k_addb<<<(L_SEQ * DMODEL + 255) / 256, 256, 0, stream>>>(seq, temb, L_SEQ);

  // --- scan orderings (hoisted: same coords every layer) ---
  k_sort<<<4, 512, 0, stream>>>(icoords, qcoords, idxb);

  // --- blocks ---
  for (int l = 0; l < NLAYERS; ++l) {
    int B0 = 24 + 20 * l;
    const float* A_log = P(B0 + 0);
    const float* Dp    = P(B0 + 1);
    const float* W_B   = P(B0 + 2);
    const float* W_C   = P(B0 + 3);
    const float* dt_b  = P(B0 + 4);
    const float* dt_w  = P(B0 + 5);
    const float* f1b   = P(B0 + 6);
    const float* f1w   = P(B0 + 7);
    const float* f2b   = P(B0 + 8);
    const float* f2w   = P(B0 + 9);
    const float* ln_b  = P(B0 + 10);
    const float* ln_g  = P(B0 + 11);
    const float* m1b   = P(B0 + 12);
    const float* m1w   = P(B0 + 13);
    const float* m2b   = P(B0 + 14);
    const float* m2w   = P(B0 + 15);
    const float* pib   = P(B0 + 16);
    const float* piw   = P(B0 + 17);
    const float* pob   = P(B0 + 18);
    const float* pow_  = P(B0 + 19);

    gemm(stream, seq, piw, pib, xin, L_SEQ, DINNER, DMODEL, ACT_NONE);          // proj_in
    gemm(stream, xin, dt_w, dt_b, dtb, L_SEQ, DINNER, DINNER, ACT_SOFTPLUS);    // dt (shared across dirs)
    gemm(stream, xin, W_B, nullptr, Bm, L_SEQ, DSTATE, DINNER, ACT_NONE);       // B
    gemm(stream, xin, W_C, nullptr, Cm, L_SEQ, DSTATE, DINNER, ACT_NONE);       // C
    k_ssm<<<dim3(DINNER / 16, 4), 256, 0, stream>>>(xin, dtb, Bm, Cm, A_log, Dp, idxb, yf);
    gemm(stream, yf, f1w, f1b, f1, L_SEQ, 2 * DINNER, 4 * DINNER, ACT_GELU);    // fusion1 + gelu
    gemm(stream, f1, f2w, f2b, f2, L_SEQ, DINNER, 2 * DINNER, ACT_NONE);        // fusion2
    gemm(stream, f2, pow_, pob, t256, L_SEQ, DMODEL, DINNER, ACT_NONE);         // proj_out
    k_add<<<(L_SEQ * DMODEL + 255) / 256, 256, 0, stream>>>(seq, t256, L_SEQ * DMODEL);
    k_ln<<<L_SEQ, 256, 0, stream>>>(seq, ln_g, ln_b, hln, DMODEL);
    gemm(stream, hln, m1w, m1b, mh, L_SEQ, 4 * DMODEL, DMODEL, ACT_GELU);       // mlp1 + gelu
    gemm(stream, mh, m2w, m2b, t256, L_SEQ, DMODEL, 4 * DMODEL, ACT_NONE);      // mlp2
    k_add<<<(L_SEQ * DMODEL + 255) / 256, 256, 0, stream>>>(seq, t256, L_SEQ * DMODEL);
  }

  // --- cross attention: queries = seq[N_IN:], kv = seq[:N_IN] ---
  const float* qtok = seq + (size_t)N_IN * DMODEL;
  gemm(stream, qtok, P(10), P(9), qb, N_OUT, DMODEL, DMODEL, ACT_NONE);   // attn_q
  gemm(stream, seq, P(6), P(5), kb, N_IN, DMODEL, DMODEL, ACT_NONE);      // attn_k
  gemm(stream, seq, P(12), P(11), vb, N_IN, DMODEL, DMODEL, ACT_NONE);    // attn_v
  k_attn<<<dim3(N_OUT, NHEADS), 128, 0, stream>>>(qb, kb, vb, ao);
  gemm(stream, ao, P(8), P(7), attn, N_OUT, DMODEL, DMODEL, ACT_NONE);    // attn_o

  // --- decoder ---
  k_ln<<<N_OUT, 256, 0, stream>>>(attn, P(20), P(19), hln, DMODEL);
  gemm(stream, hln, P(14), P(13), dh1, N_OUT, 2 * DMODEL, DMODEL, ACT_GELU);   // dec1 + gelu
  gemm(stream, dh1, P(16), P(15), t256, N_OUT, DMODEL, 2 * DMODEL, ACT_GELU);  // dec2 + gelu
  gemm(stream, t256, P(18), P(17), (float*)d_out, N_OUT, 3, DMODEL, ACT_NONE); // dec3 -> output
}
